// ResNet50DeformConv5ROIFeatureExtractor_10127532884582
// MI455X (gfx1250) — compile-verified, matrix operand order checked by reading, not run on hardware
//
#include <hip/hip_runtime.h>
#include <hip/hip_bf16.h>
#include <math.h>

// ---------------------------------------------------------------- constants
#define RP   512          // number of ROIs
#define CC   256          // channels
#define HH   96
#define WW   96
#define PP   7
#define FCN  1024
#define DIN  (CC*PP*PP)   // 12544
#define SCALE_F 0.0625f
#define GAMMA_F 0.1f

typedef __attribute__((ext_vector_type(16))) __bf16 v16bf;
typedef __attribute__((ext_vector_type(8)))  __bf16 v8bf;
typedef __attribute__((ext_vector_type(8)))  float  v8f;
typedef __attribute__((ext_vector_type(4)))  int    v4i;

// gfx1250 async global->LDS copy path (ASYNCcnt-tracked), guarded so the
// source still compiles if this toolchain doesn't declare the builtins.
#if defined(__HIP_DEVICE_COMPILE__) && \
    __has_builtin(__builtin_amdgcn_global_load_async_to_lds_b128) && \
    __has_builtin(__builtin_amdgcn_s_wait_asynccnt)
#define ASYNC_COPY 1
#else
#define ASYNC_COPY 0
#endif

// builtin signature (per hipcc diagnostic): (AS1 int4*, AS3 int4*, imm, imm)
typedef __attribute__((address_space(1))) v4i as1_v4i;
typedef __attribute__((address_space(3))) v4i as3_v4i;

// ---------------------------------------------------------------- bilinear
__device__ __forceinline__ float bilin_sample(const float* __restrict__ x,
                                              int b, int c, float gy, float gx) {
    bool valid = (gy > -1.0f) & (gy < (float)HH) & (gx > -1.0f) & (gx < (float)WW);
    float y  = fminf(fmaxf(gy, 0.0f), (float)(HH - 1));
    float xx = fminf(fmaxf(gx, 0.0f), (float)(WW - 1));
    int y0 = (int)floorf(y);
    int x0 = (int)floorf(xx);
    int y1 = min(y0 + 1, HH - 1);
    int x1 = min(x0 + 1, WW - 1);
    float ly = y  - (float)y0, lx = xx - (float)x0;
    float hy = 1.0f - ly,      hx = 1.0f - lx;
    const float* base = x + ((size_t)b * CC + c) * (HH * WW);
    float v = base[y0 * WW + x0] * (hy * hx)
            + base[y0 * WW + x1] * (hy * lx)
            + base[y1 * WW + x0] * (ly * hx)
            + base[y1 * WW + x1] * (ly * lx);
    return valid ? v : 0.0f;
}

__device__ __forceinline__ void roi_geom(const float* __restrict__ boxes, int r,
                                         float& xs1, float& ys1,
                                         float& roi_w, float& roi_h,
                                         float& bin_w, float& bin_h) {
    xs1       = boxes[r * 4 + 0] * SCALE_F - 0.5f;
    ys1       = boxes[r * 4 + 1] * SCALE_F - 0.5f;
    float xs2 = boxes[r * 4 + 2] * SCALE_F - 0.5f;
    float ys2 = boxes[r * 4 + 3] * SCALE_F - 0.5f;
    roi_w = fmaxf(xs2 - xs1, 0.1f);
    roi_h = fmaxf(ys2 - ys1, 0.1f);
    bin_w = roi_w * (1.0f / PP);
    bin_h = roi_h * (1.0f / PP);
}

// ------------------------------------------------- pool pass 1 -> feat bf16
// grid: R*49 blocks, 256 threads (one per channel)
__global__ __launch_bounds__(256) void pool1_kernel(
    const float* __restrict__ x, const float* __restrict__ boxes,
    const int* __restrict__ roi_batch, __bf16* __restrict__ feat) {
    int idx = blockIdx.x;
    int r   = idx / (PP * PP);
    int bin = idx - r * (PP * PP);
    int ph  = bin / PP, pw = bin - ph * PP;
    int c   = threadIdx.x;
    int b   = roi_batch[r];

    float xs1, ys1, roi_w, roi_h, bin_w, bin_h;
    roi_geom(boxes, r, xs1, ys1, roi_w, roi_h, bin_w, bin_h);

    float sum = 0.0f;
#pragma unroll
    for (int sy = 0; sy < 2; ++sy)
#pragma unroll
        for (int sx = 0; sx < 2; ++sx) {
            float gy = ys1 + ((float)ph + ((float)sy + 0.5f) * 0.5f) * bin_h;
            float gx = xs1 + ((float)pw + ((float)sx + 0.5f) * 0.5f) * bin_w;
            sum += bilin_sample(x, b, c, gy, gx);
        }
    feat[(size_t)r * DIN + c * (PP * PP) + bin] = (__bf16)(sum * 0.25f);
}

// ------------------------------------------------- pool pass 2 * mask -> out
__global__ __launch_bounds__(256) void pool2_kernel(
    const float* __restrict__ x, const float* __restrict__ boxes,
    const int* __restrict__ roi_batch, const float* __restrict__ off,
    const float* __restrict__ mask, float* __restrict__ out) {
    int idx = blockIdx.x;
    int r   = idx / (PP * PP);
    int bin = idx - r * (PP * PP);
    int ph  = bin / PP, pw = bin - ph * PP;
    int c   = threadIdx.x;
    int b   = roi_batch[r];

    float xs1, ys1, roi_w, roi_h, bin_w, bin_h;
    roi_geom(boxes, r, xs1, ys1, roi_w, roi_h, bin_w, bin_h);

    float offy = off[r * (2 * PP * PP) + bin];                // off[:,0]
    float offx = off[r * (2 * PP * PP) + PP * PP + bin];      // off[:,1]
    float mval = mask[r * (PP * PP) + bin];

    float sum = 0.0f;
#pragma unroll
    for (int sy = 0; sy < 2; ++sy)
#pragma unroll
        for (int sx = 0; sx < 2; ++sx) {
            float gy = ys1 + ((float)ph + ((float)sy + 0.5f) * 0.5f) * bin_h
                     + offy * roi_h * GAMMA_F;
            float gx = xs1 + ((float)pw + ((float)sx + 0.5f) * 0.5f) * bin_w
                     + offx * roi_w * GAMMA_F;
            sum += bilin_sample(x, b, c, gy, gx);
        }
    out[((size_t)r * CC + c) * (PP * PP) + bin] = sum * 0.25f * mval;
}

// -------------------------------------------------------- WMMA bf16 GEMM
// C[M,N] = relu(A[M,K](bf16) * W[K,N](f32->bf16) + bias) -> bf16
// block: 256 threads (8 waves = 4(M) x 2(N)), block tile 128x64, K-step 32.
// Each wave owns a 32x32 slab -> 4 v_wmma_f32_16x16x32_bf16 per K-step.
// LDS: A row-major [M][K]; W transposed [N][K], so every fragment is two
// contiguous 16B chunks per lane (CDNA5 16-bit matrix layout, wave32).
// A-tile staging uses GLOBAL_LOAD_ASYNC_TO_LDS_B128 when available.
#define TM 128
#define TN 64
#define TK 32
#define LDK 40   // padded K-stride in LDS (bf16 elems): rows stay 16B aligned

__global__ __launch_bounds__(256) void gemm_bf16_relu_kernel(
    const __bf16* __restrict__ A, const float* __restrict__ Wt,
    const float* __restrict__ bias, __bf16* __restrict__ out,
    int M, int N, int K) {
    __shared__ __bf16 Alds[TM * LDK];   // 10.0 KB
    __shared__ __bf16 Blds[TN * LDK];   //  5.0 KB

    const int tid   = threadIdx.x;
    const int lane  = tid & 31;
    const int wave  = tid >> 5;
    const int waveM = wave & 3;   // 4 waves along M (32 rows each)
    const int waveN = wave >> 2;  // 2 waves along N (32 cols each)
    const int m0 = blockIdx.y * TM;
    const int n0 = blockIdx.x * TN;

    // W-tile staging: 32 rows(K) x 64 cols(N) f32, each thread 8 floats
    const int wRow = tid >> 3;
    const int wCol = (tid & 7) * 8;

    // fragment addressing (CDNA5 16-bit matrix layout, wave32)
    const int kb  = (lane >> 4) * 8;          // K sub-block for this half-wave
    const int mA0 = waveM * 32 + (lane & 15); // A rows (two 16-row subtiles)
    const int mA1 = mA0 + 16;
    const int nB0 = waveN * 32 + (lane & 15); // B cols (two 16-col subtiles)
    const int nB1 = nB0 + 16;

    v8f acc00 = {}, acc01 = {}, acc10 = {}, acc11 = {};

    for (int k0 = 0; k0 < K; k0 += TK) {
        __syncthreads();   // previous tile fully consumed

        // ---- stage A tile: 128x32 bf16 (8 KB), 2 x b128 per thread ----
#if ASYNC_COPY
#pragma unroll
        for (int i = 0; i < 2; ++i) {
            int lin = tid + i * 256;
            int row = lin >> 2;          // 0..127
            int c8  = (lin & 3) * 8;     // 0,8,16,24
            __builtin_amdgcn_global_load_async_to_lds_b128(
                (as1_v4i*)(A + (size_t)(m0 + row) * K + k0 + c8),
                (as3_v4i*)(&Alds[row * LDK + c8]), 0, 0);
        }
#else
#pragma unroll
        for (int i = 0; i < 2; ++i) {
            int lin = tid + i * 256;
            int row = lin >> 2;
            int c8  = (lin & 3) * 8;
            v8bf av = *(const v8bf*)(A + (size_t)(m0 + row) * K + k0 + c8);
            *(v8bf*)(&Alds[row * LDK + c8]) = av;
        }
#endif
        // ---- stage W tile: load 8 f32 along N, convert, store [N][K] ----
        const float* wp = Wt + (size_t)(k0 + wRow) * N + n0 + wCol;
        float4 w0 = *(const float4*)(wp);
        float4 w1 = *(const float4*)(wp + 4);
        if (k0 + TK < K)  // hint the next K-slice of the weight stream
            __builtin_prefetch(wp + (size_t)TK * N, 0, 1);
        float wv[8] = {w0.x, w0.y, w0.z, w0.w, w1.x, w1.y, w1.z, w1.w};
#pragma unroll
        for (int j = 0; j < 8; ++j)
            Blds[(wCol + j) * LDK + wRow] = (__bf16)wv[j];

#if ASYNC_COPY
        __builtin_amdgcn_s_wait_asynccnt(0);
#endif
        __syncthreads();   // tile staged

        union { v16bf v; v8bf h[2]; } a0, a1, b0, b1;
        a0.h[0] = *(const v8bf*)(&Alds[mA0 * LDK + kb]);
        a0.h[1] = *(const v8bf*)(&Alds[mA0 * LDK + kb + 16]);
        a1.h[0] = *(const v8bf*)(&Alds[mA1 * LDK + kb]);
        a1.h[1] = *(const v8bf*)(&Alds[mA1 * LDK + kb + 16]);
        b0.h[0] = *(const v8bf*)(&Blds[nB0 * LDK + kb]);
        b0.h[1] = *(const v8bf*)(&Blds[nB0 * LDK + kb + 16]);
        b1.h[0] = *(const v8bf*)(&Blds[nB1 * LDK + kb]);
        b1.h[1] = *(const v8bf*)(&Blds[nB1 * LDK + kb + 16]);

        acc00 = __builtin_amdgcn_wmma_f32_16x16x32_bf16(
                    false, a0.v, false, b0.v, (short)0, acc00, false, false);
        acc01 = __builtin_amdgcn_wmma_f32_16x16x32_bf16(
                    false, a0.v, false, b1.v, (short)0, acc01, false, false);
        acc10 = __builtin_amdgcn_wmma_f32_16x16x32_bf16(
                    false, a1.v, false, b0.v, (short)0, acc10, false, false);
        acc11 = __builtin_amdgcn_wmma_f32_16x16x32_bf16(
                    false, a1.v, false, b1.v, (short)0, acc11, false, false);
    }

    // epilogue: D vgpr i -> M = i + 8*(lane>=16), N = lane&15
    const int mB   = m0 + waveM * 32 + (lane >> 4) * 8;
    const int nOut = n0 + waveN * 32 + (lane & 15);
    const float bi0 = bias[nOut];
    const float bi1 = bias[nOut + 16];
#pragma unroll
    for (int i = 0; i < 8; ++i) {
        float v00 = fmaxf(acc00[i] + bi0, 0.0f);
        float v01 = fmaxf(acc01[i] + bi1, 0.0f);
        float v10 = fmaxf(acc10[i] + bi0, 0.0f);
        float v11 = fmaxf(acc11[i] + bi1, 0.0f);
        out[(size_t)(mB + i)      * N + nOut]      = (__bf16)v00;
        out[(size_t)(mB + i)      * N + nOut + 16] = (__bf16)v01;
        out[(size_t)(mB + 16 + i) * N + nOut]      = (__bf16)v10;
        out[(size_t)(mB + 16 + i) * N + nOut + 16] = (__bf16)v11;
    }
}

// ----------------------------------------- small fp32 head (off / mask)
// act: 0 = none, 1 = sigmoid
__global__ void small_gemm_kernel(const __bf16* __restrict__ A,
                                  const float* __restrict__ Wt,
                                  const float* __restrict__ bias,
                                  float* __restrict__ out,
                                  int K, int N, int act) {
    int idx = blockIdx.x * blockDim.x + threadIdx.x;
    if (idx >= RP * N) return;
    int r = idx / N, j = idx - r * N;
    float s = bias[j];
    const __bf16* ar = A + (size_t)r * K;
    for (int k = 0; k < K; ++k)
        s += (float)ar[k] * Wt[(size_t)k * N + j];
    if (act == 1) s = 1.0f / (1.0f + expf(-s));
    out[idx] = s;
}

// ---------------------------------------------------------------- launch
extern "C" void kernel_launch(void* const* d_in, const int* in_sizes, int n_in,
                              void* d_out, int out_size, void* d_ws, size_t ws_size,
                              hipStream_t stream) {
    const float* x         = (const float*)d_in[0];
    const float* boxes     = (const float*)d_in[1];
    const int*   roi_batch = (const int*)d_in[2];
    const float* W1  = (const float*)d_in[3];
    const float* b1  = (const float*)d_in[4];
    const float* W2  = (const float*)d_in[5];
    const float* b2  = (const float*)d_in[6];
    const float* W3  = (const float*)d_in[7];
    const float* b3  = (const float*)d_in[8];
    const float* M1  = (const float*)d_in[9];
    const float* mb1 = (const float*)d_in[10];
    const float* M2  = (const float*)d_in[11];
    const float* mb2 = (const float*)d_in[12];

    char* ws = (char*)d_ws;
    size_t o = 0;
    __bf16* feat = (__bf16*)(ws + o); o += (size_t)RP * DIN * 2;   // 12.85 MB
    __bf16* h1   = (__bf16*)(ws + o); o += (size_t)RP * FCN * 2;   // 1 MB
    __bf16* m1   = (__bf16*)(ws + o); o += (size_t)RP * FCN * 2;   // 1 MB
    __bf16* h2   = (__bf16*)(ws + o); o += (size_t)RP * FCN * 2;   // 1 MB
    float*  offb = (float*)(ws + o);  o += (size_t)RP * 2 * PP * PP * 4;
    float*  mskb = (float*)(ws + o);  o += (size_t)RP * PP * PP * 4;

    const int nBins = RP * PP * PP;  // 25088 blocks

    // 1) deform pool (zero offsets) -> feat (bf16)
    pool1_kernel<<<nBins, 256, 0, stream>>>(x, boxes, roi_batch, feat);

    // 2) big bf16 WMMA GEMMs (weights stream once from HBM, then 192MB L2)
    dim3 gBig(FCN / TN, RP / TM);  // 16 x 4
    gemm_bf16_relu_kernel<<<gBig, 256, 0, stream>>>(feat, W1, b1, h1, RP, FCN, DIN);
    gemm_bf16_relu_kernel<<<gBig, 256, 0, stream>>>(feat, M1, mb1, m1, RP, FCN, DIN);
    gemm_bf16_relu_kernel<<<gBig, 256, 0, stream>>>(h1, W2, b2, h2, RP, FCN, FCN);

    // 3) small fp32 heads: offsets (linear) and mask (sigmoid)
    int nOff = RP * 2 * PP * PP;   // 512*98
    int nMsk = RP * PP * PP;       // 512*49
    small_gemm_kernel<<<(nOff + 255) / 256, 256, 0, stream>>>(h2, W3, b3, offb,
                                                              FCN, 2 * PP * PP, 0);
    small_gemm_kernel<<<(nMsk + 255) / 256, 256, 0, stream>>>(m1, M2, mb2, mskb,
                                                              FCN, PP * PP, 1);

    // 4) deform pool with predicted offsets, scaled by mask -> output (f32)
    pool2_kernel<<<nBins, 256, 0, stream>>>(x, boxes, roi_batch, offb, mskb,
                                            (float*)d_out);
}
